// PuriAttention_25537875542176
// MI455X (gfx1250) — compile-verified, matrix operand order checked by reading, not run on hardware
//
#include <hip/hip_runtime.h>
#include <hip/hip_bf16.h>

typedef __attribute__((ext_vector_type(16))) _Float16 v16h;
typedef __attribute__((ext_vector_type(8)))  float    v8f;

#define B_  8
#define S_  4096
#define H_  1024
#define NH_ 16
#define HD_ 64
#define SCALE 0.125f  // 1/sqrt(64)

// ---------------------------------------------------------------------------
// Kernel 1: scores[b,h,s] = (K[b,s,h,:] . q[b,0,h,:]) * SCALE + mask
// One wave handles one 16-row s-tile for one (b,h). Depth 64 = two
// v_wmma_f32_16x16x32_f16 with f32 accumulation. B-matrix = q broadcast
// into all 16 columns, so every column of D equals the 16 scores.
// ---------------------------------------------------------------------------
__global__ __launch_bounds__(256)
void puri_scores_wmma(const float* __restrict__ q,
                      const float* __restrict__ k,
                      const float* __restrict__ am,
                      float* __restrict__ scores) {
    const int lane = threadIdx.x & 31;
    const int wid  = threadIdx.x >> 5;
    const int tile = blockIdx.x * 8 + wid;        // 32768 tiles total
    const int bh   = tile >> 8;                   // 256 tiles per (b,h)
    const int st   = tile & 255;
    const int b    = bh >> 4;
    const int h    = bh & 15;
    const int s_base = st << 4;

    // Per-lane K-depth mapping for 16-bit A (16x32): lanes 0-15 take K base 0,
    // lanes 16-31 take K base 8; elements 0..7 -> K+0..7, 8..15 -> K+16..23.
    const int kbase = (lane < 16) ? 0 : 8;

    // ---- B matrix: q vector for this head, broadcast across all columns ----
    const float* qp = q + ((size_t)b * S_) * H_ + h * HD_;  // q[b,0,h,:]
    v16h b0, b1;
    #pragma unroll
    for (int e = 0; e < 8; ++e) {
        b0[e]     = (_Float16)qp[kbase + e];
        b0[e + 8] = (_Float16)qp[kbase + 16 + e];
        b1[e]     = (_Float16)qp[kbase + 32 + e];
        b1[e + 8] = (_Float16)qp[kbase + 48 + e];
    }

    // ---- A matrix: 16 rows of K (s positions), 64 deep -> two 16x32 tiles --
    const int   srow = s_base + (lane & 15);
    const float* kp  = k + (((size_t)b * S_ + srow) * NH_ + h) * HD_;
    v16h a0, a1;
    #pragma unroll
    for (int e = 0; e < 8; ++e) {
        a0[e]     = (_Float16)kp[kbase + e];
        a0[e + 8] = (_Float16)kp[kbase + 16 + e];
        a1[e]     = (_Float16)kp[kbase + 32 + e];
        a1[e + 8] = (_Float16)kp[kbase + 48 + e];
    }

    v8f c = {};
    c = __builtin_amdgcn_wmma_f32_16x16x32_f16(false, a0, false, b0,
                                               (short)0, c, false, false);
    c = __builtin_amdgcn_wmma_f32_16x16x32_f16(false, a1, false, b1,
                                               (short)0, c, false, false);

    // D layout: VGPR r, lanes 0-15 -> M=r, lanes 16-31 -> M=8+r. All columns
    // identical, so lanes with column N==0 (lane 0 and lane 16) emit results.
    if ((lane & 15) == 0) {
        const int m_off = (lane >> 4) << 3;
        float* srow_out = scores + (size_t)bh * S_ + s_base + m_off;
        const float* amrow = am + (size_t)b * S_ + s_base + m_off;
        #pragma unroll
        for (int r = 0; r < 8; ++r) {
            int s = s_base + m_off + r;
            float msk = amrow[r] + ((s == 0) ? -10000.0f : 0.0f);
            srow_out[r] = c[r] * SCALE + msk;
        }
    }
}

// ---------------------------------------------------------------------------
// Kernel 2: row softmax over S=4096 per (b,h). 128 blocks x 256 threads.
// ---------------------------------------------------------------------------
__global__ __launch_bounds__(256)
void puri_softmax(const float* __restrict__ scores, float* __restrict__ probs) {
    __shared__ float red[256];
    const int bh = blockIdx.x;
    const int t  = threadIdx.x;
    const float* row  = scores + (size_t)bh * S_;
    float*       prow = probs  + (size_t)bh * S_;

    float m = -3.0e38f;
    for (int s = t; s < S_; s += 256) m = fmaxf(m, row[s]);
    red[t] = m; __syncthreads();
    for (int off = 128; off > 0; off >>= 1) {
        if (t < off) red[t] = fmaxf(red[t], red[t + off]);
        __syncthreads();
    }
    m = red[0]; __syncthreads();

    float sum = 0.0f;
    for (int s = t; s < S_; s += 256) {
        float e = __expf(row[s] - m);
        prow[s] = e;
        sum += e;
    }
    red[t] = sum; __syncthreads();
    for (int off = 128; off > 0; off >>= 1) {
        if (t < off) red[t] += red[t + off];
        __syncthreads();
    }
    const float inv = 1.0f / red[0];
    for (int s = t; s < S_; s += 256) prow[s] *= inv;
}

// ---------------------------------------------------------------------------
// Kernel 3: ctx[b,h,d] = sum_s probs[b,h,s] * V[b,s,h,d].
// 128 blocks (one per (b,h)), 256 threads: 4 s-groups x 64 d-lanes.
// V rows are 64 contiguous floats -> coalesced streaming of the full 128 MiB.
// ---------------------------------------------------------------------------
__global__ __launch_bounds__(256)
void puri_ctx(const float* __restrict__ probs,
              const float* __restrict__ v,
              float* __restrict__ ctx) {
    const int bh = blockIdx.x;
    const int b  = bh >> 4;
    const int h  = bh & 15;
    const int t  = threadIdx.x;
    const int d  = t & 63;
    const int sg = t >> 6;                 // 0..3

    const float* pv = v + ((size_t)b * S_ * NH_ + h) * HD_;  // + s*NH_*HD_ + d
    const float* pp = probs + (size_t)bh * S_;

    float acc = 0.0f;
    for (int s = sg; s < S_; s += 4)
        acc = fmaf(pp[s], pv[(size_t)s * (NH_ * HD_) + d], acc);

    __shared__ float sh[4][HD_];
    sh[sg][d] = acc;
    __syncthreads();
    if (t < HD_)
        ctx[(size_t)b * H_ + h * HD_ + d] =
            sh[0][d] + sh[1][d] + sh[2][d] + sh[3][d];
}

extern "C" void kernel_launch(void* const* d_in, const int* in_sizes, int n_in,
                              void* d_out, int out_size, void* d_ws, size_t ws_size,
                              hipStream_t stream) {
    const float* q  = (const float*)d_in[0];
    const float* k  = (const float*)d_in[1];
    const float* v  = (const float*)d_in[2];
    const float* am = (const float*)d_in[3];

    float* out    = (float*)d_out;
    float* attn   = out;                       // (B,1,H)      = 8192
    float* scores = out + (size_t)B_ * H_;     // (B,1,NH,S)   = 524288
    float* probs  = scores + (size_t)B_ * NH_ * S_;

    // 8*16*(4096/16) = 32768 wave-tiles, 8 waves per 256-thread block
    puri_scores_wmma<<<4096, 256, 0, stream>>>(q, k, am, scores);
    puri_softmax   <<<B_ * NH_, 256, 0, stream>>>(scores, probs);
    puri_ctx       <<<B_ * NH_, 256, 0, stream>>>(probs, v, attn);
}
